// S4LM_25039659335761
// MI455X (gfx1250) — matrix operand order repaired and checked by reading, compiled, not run on hardware
//
#include <hip/hip_runtime.h>
#include <hip/hip_bf16.h>
#include <math.h>

// ---- model dims (fixed by reference) ----
#define BSZ   2
#define SEQ   2048
#define DM    512
#define NLAY  4
#define NST   32
#define VOC   50257
#define VPAD  50304              // ceil(VOC/64)*64 (also mult of 128)
#define MROWS (BSZ*SEQ)          // 4096

typedef __bf16 bf16_t;
typedef bf16_t v16bf __attribute__((ext_vector_type(16)));
typedef bf16_t v8bf  __attribute__((ext_vector_type(8)));
typedef float  v8f   __attribute__((ext_vector_type(8)));

// ---------------------------------------------------------------------------
// CDNA5 async copy: 16B global -> LDS, per-lane LDS dest, tracked by ASYNCcnt.
__device__ __forceinline__ void async_b128(unsigned lds_off, const void* gptr) {
    asm volatile("global_load_async_to_lds_b128 %0, %1, off"
                 :: "v"(lds_off), "v"(gptr) : "memory");
}
__device__ __forceinline__ unsigned lds_addr(const void* p) {
    // flat LDS aperture keeps the byte offset in addr[31:0] (ISA 10.2)
    return (unsigned)(unsigned long long)p;
}
__device__ __forceinline__ void wait_async0() {
    asm volatile("s_wait_asynccnt 0x0" ::: "memory");
}

// ---------------------------------------------------------------------------
// f32 -> bf16 elementwise convert
__global__ void s4lm_cvt_bf16(const float* __restrict__ in, bf16_t* __restrict__ out, int n) {
    int i = blockIdx.x * 256 + threadIdx.x;
    if (i < n) out[i] = (bf16_t)in[i];
}

// head_w f32 (VOC,512) -> bf16 (VPAD,512), zero-padded rows
__global__ void s4lm_cvt_head(const float* __restrict__ in, bf16_t* __restrict__ out) {
    long long i = (long long)blockIdx.x * 256 + threadIdx.x;
    if (i >= (long long)VPAD * DM) return;
    int row = (int)(i >> 9);
    int col = (int)(i & 511);
    out[i] = (row < VOC) ? (bf16_t)in[(long long)row * DM + col] : (bf16_t)0.0f;
}

// ---------------------------------------------------------------------------
// x[b,t,:] = token_embed[ids[b,t],:] + pos_embed[t,:]
__global__ void s4lm_embed(const int* __restrict__ ids, const float* __restrict__ tok,
                           const float* __restrict__ pos, float* __restrict__ X) {
    int i = blockIdx.x * 256 + threadIdx.x;          // over MROWS*DM
    if (i >= MROWS * DM) return;
    int m = i >> 9;                                  // row (b*SEQ + t)
    int d = i & 511;
    int t = m & (SEQ - 1);
    X[i] = tok[(long long)ids[m] * DM + d] + pos[(long long)t * DM + d];
}

// ---------------------------------------------------------------------------
// S4D causal conv as diagonal recurrence + skip + GELU.  One wave per (b,d)
// channel; lane n owns complex mode n (N=32 == wave32 lanes).
__global__ __launch_bounds__(32)
void s4lm_scan(const float* __restrict__ X,        // (MROWS, DM)
               const float* __restrict__ log_dt,   // (DM)   this layer
               const float* __restrict__ C_re,     // (DM,N) this layer
               const float* __restrict__ C_im,
               const float* __restrict__ skipD,    // (DM)
               bf16_t* __restrict__ Yh)            // (MROWS, DM) gelu(y) in bf16
{
    const int ch = blockIdx.x;                      // 0 .. BSZ*DM-1
    const int b  = ch / DM;
    const int d  = ch - b * DM;
    const int n  = threadIdx.x;                     // mode index

    const float pi = 3.14159265358979323846f;
    float dt  = __expf(log_dt[d]);
    float mag = __expf(-0.5f * dt);
    float th  = dt * pi * (float)n;
    float lr  = mag * __cosf(th);                   // lambda = exp(dt*A)
    float li  = mag * __sinf(th);
    // C' = (C_re + i C_im) * (lambda - 1) / A,  A = -0.5 + i*pi*n
    float ar = -0.5f, ai = pi * (float)n;
    float den = ar * ar + ai * ai;
    float dr = lr - 1.0f, di = li;
    float qr = (dr * ar + di * ai) / den;
    float qi = (di * ar - dr * ai) / den;
    float c0r = C_re[d * NST + n], c0i = C_im[d * NST + n];
    float cr = c0r * qr - c0i * qi;
    float ci = c0r * qi + c0i * qr;
    float sd = skipD[d];

    const float* xp = X  + (long long)b * SEQ * DM + d;
    bf16_t*      yp = Yh + (long long)b * SEQ * DM + d;

    float sr = 0.0f, si = 0.0f;
    for (int t = 0; t < SEQ; ++t) {
        float u   = xp[(long long)t * DM];
        float nsr = lr * sr - li * si + u;          // s = lambda*s + u
        float nsi = lr * si + li * sr;
        sr = nsr; si = nsi;
        float c = cr * sr - ci * si;                // Re(C' * s)
        #pragma unroll
        for (int m = 16; m > 0; m >>= 1) c += __shfl_xor(c, m, 32);
        float y  = 2.0f * c + sd * u;
        float y3 = y * y * y;                       // tanh-approx GELU (jax default)
        float g  = 0.5f * y * (1.0f + tanhf(0.7978845608f * (y + 0.044715f * y3)));
        if (n == 0) yp[(long long)t * DM] = (bf16_t)g;
    }
}

// ---------------------------------------------------------------------------
// GLU output projection + residual:  X += z[:,:D] * sigmoid(z[:,D:]),
// z = Yh @ Wo^T + b.  One wave per block; 16M x 64N tile, lo+hi halves fused.
// Wo (1 MB/layer) is L2/L0-resident so direct global streaming is fine here.
__global__ __launch_bounds__(32)
void s4lm_glu_proj(const bf16_t* __restrict__ Yh,   // (MROWS, 512) bf16
                   const bf16_t* __restrict__ Wo,   // (1024, 512)  bf16, this layer
                   const float*  __restrict__ bias, // (1024)       this layer
                   float* __restrict__ X)           // (MROWS, 512) residual in/out
{
    const int K    = DM;
    const int lane = threadIdx.x;
    const int row  = lane & 15;
    const int hi   = lane >> 4;
    const int m0   = blockIdx.x * 16;
    const int n0   = blockIdx.y * 64;               // GLU-lo column block; hi at +512

    v8f acc[8] = {};                                // [0..3]=z_lo tiles, [4..7]=z_hi
    const bf16_t* Ap = Yh + (long long)(m0 + row) * K + hi * 16;

    for (int k = 0; k < K; k += 32) {
        __builtin_prefetch((const void*)(Ap + k + 64), 0, 1);
        v8bf alo = *(const v8bf*)(Ap + k);
        v8bf ahi = *(const v8bf*)(Ap + k + 8);
        v16bf a;
        #pragma unroll
        for (int i = 0; i < 8; ++i) { a[i] = alo[i]; a[8 + i] = ahi[i]; }
        #pragma unroll
        for (int j = 0; j < 4; ++j) {
            const bf16_t* Bl = Wo + (long long)(n0 + j * 16 + row) * K + hi * 16 + k;
            const bf16_t* Bh = Bl + (long long)DM * K;   // hi-half rows (+512)
            v16bf bl = *(const v16bf*)Bl;
            v16bf bh = *(const v16bf*)Bh;
            acc[j]     = __builtin_amdgcn_wmma_f32_16x16x32_bf16(
                             false, a, false, bl, (short)0, acc[j],     false, false);
            acc[4 + j] = __builtin_amdgcn_wmma_f32_16x16x32_bf16(
                             false, a, false, bh, (short)0, acc[4 + j], false, false);
        }
    }

    const int ncol = lane & 15;
    #pragma unroll
    for (int j = 0; j < 4; ++j) {
        int n  = n0 + j * 16 + ncol;
        float bl = bias[n], bh = bias[n + DM];
        #pragma unroll
        for (int r = 0; r < 8; ++r) {
            int   m  = m0 + r + 8 * hi;             // C/D layout: vgpr r, lane-half
            float zl = acc[j][r] + bl;
            float zh = acc[4 + j][r] + bh;
            float g  = zl / (1.0f + __expf(-zh));   // zl * sigmoid(zh)
            X[(long long)m * DM + n] += g;
        }
    }
}

// ---------------------------------------------------------------------------
// LayerNorm over D=512 -> bf16
__global__ __launch_bounds__(256)
void s4lm_layernorm(const float* __restrict__ X, const float* __restrict__ g,
                    const float* __restrict__ bt, bf16_t* __restrict__ Xh)
{
    int mrow = blockIdx.x;
    int tid  = threadIdx.x;
    const float* xr = X + (long long)mrow * DM;
    float v0 = xr[tid], v1 = xr[tid + 256];
    float s = v0 + v1, ss = v0 * v0 + v1 * v1;
    #pragma unroll
    for (int m = 16; m > 0; m >>= 1) { s += __shfl_xor(s, m, 32); ss += __shfl_xor(ss, m, 32); }
    __shared__ float sb[8], ssb[8], mu_s, rs_s;
    int wid = tid >> 5;
    if ((tid & 31) == 0) { sb[wid] = s; ssb[wid] = ss; }
    __syncthreads();
    if (tid == 0) {
        float S = 0.0f, SS = 0.0f;
        #pragma unroll
        for (int i = 0; i < 8; ++i) { S += sb[i]; SS += ssb[i]; }
        float mu  = S / (float)DM;
        float var = SS / (float)DM - mu * mu;
        mu_s = mu; rs_s = rsqrtf(var + 1e-5f);
    }
    __syncthreads();
    float mu = mu_s, rs = rs_s;
    Xh[(long long)mrow * DM + tid]       = (bf16_t)((v0 - mu) * rs * g[tid]       + bt[tid]);
    Xh[(long long)mrow * DM + tid + 256] = (bf16_t)((v1 - mu) * rs * g[tid + 256] + bt[tid + 256]);
}

// ---------------------------------------------------------------------------
// Head GEMM: logits = Xh @ head_w^T.
// 256-thread WG (8 waves) computes a 128M x 128N macro-tile; K streamed in
// 32-wide chunks through double-buffered LDS via global_load_async_to_lds_b128
// (ASYNCcnt), overlapping chunk c+1 loads with chunk c WMMAs.  LDS rows padded
// 64B->80B to kill the 16-lane bank conflict on fragment reads.
#define KC         32           // K halves per chunk
#define LDSW       40           // halves per LDS row (32 + 8 pad)
#define NCHUNK     (DM / KC)    // 16

__global__ __launch_bounds__(256)
void s4lm_head(const bf16_t* __restrict__ Xh,      // (MROWS, 512) bf16
               const bf16_t* __restrict__ Wh,      // (VPAD, 512)  bf16
               float* __restrict__ Out)            // (MROWS, VOC) f32
{
    __shared__ bf16_t sA[2][128 * LDSW];           // 2 x 10 KB
    __shared__ bf16_t sB[2][128 * LDSW];           // 2 x 10 KB

    const int tid  = threadIdx.x;
    const int lane = tid & 31;
    const int wid  = tid >> 5;
    const int wm   = wid & 3;                      // wave M index (0..3)
    const int wn   = wid >> 2;                     // wave N index (0..1)
    const int r    = lane & 15;
    const int hi   = lane >> 4;
    const int mBase = blockIdx.x * 128;
    const int nBase = blockIdx.y * 128;

    const bf16_t* Ag = Xh + (long long)mBase * DM;
    const bf16_t* Bg = Wh + (long long)nBase * DM;

    v8f acc[2][4] = {};                            // 32M x 64N per wave

    auto stage = [&](int buf, int k0) {            // 16 KB: 2 A + 2 B asyncs/thread
        #pragma unroll
        for (int s = tid; s < 512; s += 256) {
            int row  = s >> 2;                     // 0..127
            int part = s & 3;                      // 16B sub-segment
            async_b128(lds_addr(&sA[buf][row * LDSW + part * 8]),
                       Ag + (long long)row * DM + k0 + part * 8);
            async_b128(lds_addr(&sB[buf][row * LDSW + part * 8]),
                       Bg + (long long)row * DM + k0 + part * 8);
        }
    };

    stage(0, 0);
    wait_async0();
    __syncthreads();

    for (int c = 0; c < NCHUNK; ++c) {
        int buf = c & 1;
        if (c + 1 < NCHUNK) stage(buf ^ 1, (c + 1) * KC);

        v16bf afrag[2];
        #pragma unroll
        for (int i = 0; i < 2; ++i) {              // A rows: wm*32 + i*16 + r
            const bf16_t* ap = &sA[buf][(wm * 32 + i * 16 + r) * LDSW + hi * 16];
            v8bf lo = *(const v8bf*)ap;            // K [16*hi,   16*hi+8)
            v8bf hh = *(const v8bf*)(ap + 8);      // K [16*hi+8, 16*hi+16)
            #pragma unroll
            for (int e = 0; e < 8; ++e) { afrag[i][e] = lo[e]; afrag[i][8 + e] = hh[e]; }
        }
        #pragma unroll
        for (int j = 0; j < 4; ++j) {              // B rows: wn*64 + j*16 + r
            const bf16_t* bp = &sB[buf][(wn * 64 + j * 16 + r) * LDSW + hi * 16];
            v8bf lo = *(const v8bf*)bp;            // K [16*hi, 16*hi+16)
            v8bf hh = *(const v8bf*)(bp + 8);
            v16bf bfrag;
            #pragma unroll
            for (int e = 0; e < 8; ++e) { bfrag[e] = lo[e]; bfrag[8 + e] = hh[e]; }
            #pragma unroll
            for (int i = 0; i < 2; ++i)
                acc[i][j] = __builtin_amdgcn_wmma_f32_16x16x32_bf16(
                                false, afrag[i], false, bfrag, (short)0, acc[i][j],
                                false, false);
        }
        wait_async0();                             // next buffer landed
        __syncthreads();                           // this buffer fully consumed
    }

    #pragma unroll
    for (int i = 0; i < 2; ++i) {
        #pragma unroll
        for (int j = 0; j < 4; ++j) {
            int n = nBase + wn * 64 + j * 16 + r;
            if (n < VOC) {
                #pragma unroll
                for (int rr = 0; rr < 8; ++rr) {
                    int m = mBase + wm * 32 + i * 16 + rr + 8 * hi;
                    Out[(long long)m * VOC + n] = acc[i][j][rr];
                }
            }
        }
    }
}

// ---------------------------------------------------------------------------
extern "C" void kernel_launch(void* const* d_in, const int* in_sizes, int n_in,
                              void* d_out, int out_size, void* d_ws, size_t ws_size,
                              hipStream_t stream) {
    const int*   ids    = (const int*)  d_in[0];
    const float* tok    = (const float*)d_in[1];
    const float* pos    = (const float*)d_in[2];
    const float* log_dt = (const float*)d_in[3];
    const float* C_re   = (const float*)d_in[4];
    const float* C_im   = (const float*)d_in[5];
    const float* skipD  = (const float*)d_in[6];
    const float* out_w  = (const float*)d_in[7];
    const float* out_b  = (const float*)d_in[8];
    const float* ln_g   = (const float*)d_in[9];
    const float* ln_b   = (const float*)d_in[10];
    const float* head_w = (const float*)d_in[11];
    float* logits = (float*)d_out;

    // workspace layout
    char* ws = (char*)d_ws;
    float*  X   = (float*) (ws);                                  // 8 MB  f32 (MROWS,512)
    bf16_t* Yh  = (bf16_t*)(ws + (size_t)8  * 1024 * 1024);       // 4 MB  bf16 (MROWS,512)
    bf16_t* Xh  = (bf16_t*)(ws + (size_t)12 * 1024 * 1024);       // 4 MB  bf16 (MROWS,512)
    bf16_t* Wo  = (bf16_t*)(ws + (size_t)16 * 1024 * 1024);       // 4 MB  bf16 (NLAY,1024,512)
    bf16_t* Whd = (bf16_t*)(ws + (size_t)20 * 1024 * 1024);       // 51.5 MB bf16 (VPAD,512)
    (void)ws_size; (void)in_sizes; (void)n_in; (void)out_size;    // ~72 MB total needed

    // 1. weight conversions (every call; deterministic, capture-safe)
    {
        int n = NLAY * 2 * DM * DM;                               // 2,097,152
        s4lm_cvt_bf16<<<(n + 255) / 256, 256, 0, stream>>>(out_w, Wo, n);
        long long nh = (long long)VPAD * DM;                      // 25,755,648
        s4lm_cvt_head<<<(unsigned)((nh + 255) / 256), 256, 0, stream>>>(head_w, Whd);
    }

    // 2. embedding
    s4lm_embed<<<(MROWS * DM + 255) / 256, 256, 0, stream>>>(ids, tok, pos, X);

    // 3. S4 layers: scan (conv+skip+gelu) then GLU projection with residual
    for (int l = 0; l < NLAY; ++l) {
        s4lm_scan<<<BSZ * DM, 32, 0, stream>>>(
            X, log_dt + (long long)l * DM,
            C_re + (long long)l * DM * NST, C_im + (long long)l * DM * NST,
            skipD + (long long)l * DM, Yh);
        dim3 g(MROWS / 16, DM / 64);                              // (256, 8)
        s4lm_glu_proj<<<g, 32, 0, stream>>>(
            Yh, Wo + (long long)l * 2 * DM * DM, out_b + (long long)l * 2 * DM, X);
    }

    // 4. final LayerNorm -> bf16
    s4lm_layernorm<<<MROWS, 256, 0, stream>>>(X, ln_g, ln_b, Xh);

    // 5. vocab head GEMM (dominant cost: ~211 GFLOP, bf16 WMMA + async LDS)
    dim3 gh(MROWS / 128, VPAD / 128);                             // (32, 393)
    s4lm_head<<<gh, 256, 0, stream>>>(Xh, Whd, logits);
}
